// S5SSM_32263794327790
// MI455X (gfx1250) — matrix operand order repaired and checked by reading, not compiled, not run on hardware
//
#include <hip/hip_runtime.h>
#include <hip/hip_bf16.h>
#include <math.h>

// ---------------- problem constants ----------------
constexpr int kL = 16384;             // sequence length
constexpr int kH = 1024;              // hidden
constexpr int kP = 512;               // state dim
constexpr int k2P = 2 * kP;           // 1024 (re/im interleaved)
constexpr int kNCHUNK = 256;          // scan chunks
constexpr int kTCHUNK = kL / kNCHUNK; // 64

// GEMM tiling: 256 threads (8 wave32), 128x64 tile, K-step 32, LDS double buffered
constexpr int BM = 128, BN = 64, BK = 32;
constexpr int LDSP = 40;              // row pitch (bf16), 80B = 16B-aligned rows
constexpr int NTHR = 256;

typedef __attribute__((ext_vector_type(16))) __bf16 v16bf;
typedef __attribute__((ext_vector_type(8)))  float  v8f;

// ---------------- CDNA5 async-to-LDS helpers ----------------
__device__ inline unsigned lds_off(const void* p) {
    // flat LDS address: low 32 bits are the wave-relative LDS offset
    return (unsigned)(unsigned long long)p;
}
__device__ inline void async_ld_b128(unsigned lds, const void* gaddr) {
    asm volatile("global_load_async_to_lds_b128 %0, %1, off"
                 :: "v"(lds), "v"((unsigned long long)gaddr) : "memory");
}
__device__ inline void wait_async0() {
    asm volatile("s_wait_asynccnt 0x0" ::: "memory");
}

// ==================== stage 0: discretization ====================
__global__ void k_prep_lambda(const float* __restrict__ Lre,
                              const float* __restrict__ Lim,
                              const float* __restrict__ log_step,
                              float* __restrict__ lb) {
    int p = threadIdx.x;                       // blockDim = kP
    float step = expf(log_step[p]);
    float ar = Lre[p] * step, ai = Lim[p] * step;
    float er = expf(ar);
    lb[2 * p]     = er * cosf(ai);
    lb[2 * p + 1] = er * sinf(ai);
}

// W1t (2P x H), K(=h)-contiguous rows: W1t[2p+j][h] = re/im(((lb-1)/Lambda)*B_tilde[p,h])
__global__ void k_prep_W1(const float* __restrict__ Lre, const float* __restrict__ Lim,
                          const float* __restrict__ B, const float* __restrict__ lb,
                          __bf16* __restrict__ W1t) {
    int idx = blockIdx.x * blockDim.x + threadIdx.x;   // over P*H, h fastest
    int h = idx % kH, p = idx / kH;
    float lr = Lre[p], li = Lim[p];
    float inv = 1.0f / (lr * lr + li * li);
    float dr = lb[2 * p] - 1.0f, di = lb[2 * p + 1];
    float cr = (dr * lr + di * li) * inv;
    float ci = (di * lr - dr * li) * inv;
    float br = B[(size_t)(p * kH + h) * 2 + 0];
    float bi = B[(size_t)(p * kH + h) * 2 + 1];
    W1t[(size_t)(2 * p) * kH + h]     = (__bf16)(cr * br - ci * bi);
    W1t[(size_t)(2 * p + 1) * kH + h] = (__bf16)(cr * bi + ci * br);
}

// W2t (H x 2P), K(=2p)-contiguous rows: W2t[h][2p]=2*C_re[h,p], W2t[h][2p+1]=-2*C_im[h,p]
__global__ void k_prep_W2(const float* __restrict__ C, __bf16* __restrict__ W2t) {
    int idx = blockIdx.x * blockDim.x + threadIdx.x;   // over H*P, p fastest
    int p = idx % kP, h = idx / kP;
    float cr = C[(size_t)(h * kP + p) * 2 + 0];
    float ci = C[(size_t)(h * kP + p) * 2 + 1];
    W2t[(size_t)h * k2P + 2 * p]     = (__bf16)(2.0f * cr);
    W2t[(size_t)h * k2P + 2 * p + 1] = (__bf16)(-2.0f * ci);
}

// ==================== WMMA fragment helpers ====================
// A-frag (16x32): lanes 0-15 row=lane K{0..7,16..23}; lanes 16-31 K{8..15,24..31}
__device__ inline v16bf load_afrag(const __bf16* arow, int half) {
    v16bf a;
#pragma unroll
    for (int i = 0; i < 8; ++i) a[i] = arow[half * 8 + i];
#pragma unroll
    for (int i = 0; i < 8; ++i) a[8 + i] = arow[16 + half * 8 + i];
    return a;
}
// B-frag (32x16): lanes 0-15 col=lane K{0..7,16..23}; lanes 16-31 K{8..15,24..31} (Bs is [n][k])
__device__ inline v16bf load_bfrag(const __bf16* bcol, int half) {
    v16bf b;
#pragma unroll
    for (int i = 0; i < 8; ++i) b[i] = bcol[half * 8 + i];
#pragma unroll
    for (int i = 0; i < 8; ++i) b[8 + i] = bcol[16 + half * 8 + i];
    return b;
}

// ==================== GEMM-1: Bu = u(f32) @ W1t^T -> f32 (L x 2P) ====================
__global__ void __launch_bounds__(NTHR)
k_gemm1(const float* __restrict__ A, const __bf16* __restrict__ Bw,
        float* __restrict__ Cout) {
    __shared__ __bf16 As[2][BM][LDSP];
    __shared__ __bf16 Bs[2][BN][LDSP];
    const int tid = threadIdx.x, lane = tid & 31, wave = tid >> 5;
    const int half = lane >> 4, l15 = lane & 15;
    const int m0 = blockIdx.y * BM, n0 = blockIdx.x * BN;
    constexpr int K = kH, N = k2P, nK = K / BK;

    // B async staging: 1 x b128 per thread per tile (Bw is (2P x H), K-contiguous)
    const int bn = tid >> 2, bc = (tid & 3) * 8;
    const __bf16* bsrc = Bw + (size_t)(n0 + bn) * K + bc;
    const unsigned bdst[2] = { lds_off(&Bs[0][bn][bc]), lds_off(&Bs[1][bn][bc]) };

    // A register staging: 4 x float4 per thread
    int arow[4], acol[4];
#pragma unroll
    for (int it = 0; it < 4; ++it) {
        int q = tid + NTHR * it;
        arow[it] = q >> 3; acol[it] = (q & 7) * 4;
    }
    float4 ra[4];
    auto loadA = [&](int k0) {
#pragma unroll
        for (int it = 0; it < 4; ++it)
            ra[it] = *(const float4*)&A[(size_t)(m0 + arow[it]) * K + k0 + acol[it]];
    };
    auto storeA = [&](int buf) {
#pragma unroll
        for (int it = 0; it < 4; ++it) {
            As[buf][arow[it]][acol[it] + 0] = (__bf16)ra[it].x;
            As[buf][arow[it]][acol[it] + 1] = (__bf16)ra[it].y;
            As[buf][arow[it]][acol[it] + 2] = (__bf16)ra[it].z;
            As[buf][arow[it]][acol[it] + 3] = (__bf16)ra[it].w;
        }
    };

    v8f acc[4];
#pragma unroll
    for (int ns = 0; ns < 4; ++ns) acc[ns] = (v8f){0.f,0.f,0.f,0.f,0.f,0.f,0.f,0.f};

    // prologue: tile 0
    loadA(0);
    async_ld_b128(bdst[0], bsrc);
    storeA(0);
    wait_async0();
    __syncthreads();

    for (int kt = 0; kt < nK; ++kt) {
        const int buf = kt & 1;
        if (kt + 1 < nK) {                       // issue next tile early
            loadA((kt + 1) * BK);
            async_ld_b128(bdst[buf ^ 1], bsrc + (kt + 1) * BK);
        }
        v16bf af = load_afrag(&As[buf][16 * wave + l15][0], half);
#pragma unroll
        for (int ns = 0; ns < 4; ++ns) {
            v16bf bf_ = load_bfrag(&Bs[buf][16 * ns + l15][0], half);
            acc[ns] = __builtin_amdgcn_wmma_f32_16x16x32_bf16(
                false, af, false, bf_, (short)0, acc[ns], false, false);
        }
        if (kt + 1 < nK) { storeA(buf ^ 1); wait_async0(); }
        __syncthreads();
    }
#pragma unroll
    for (int ns = 0; ns < 4; ++ns)
#pragma unroll
        for (int i = 0; i < 8; ++i) {
            int r = m0 + 16 * wave + 8 * half + i;
            int c = n0 + 16 * ns + l15;
            Cout[(size_t)r * N + c] = acc[ns][i];
        }
}

// ==================== scan: x_t = a*x_{t-1} + Bu_t (complex, per p) ====================
__global__ void k_scan_local(const float* __restrict__ lb, float* __restrict__ Bu,
                             float* __restrict__ carry) {
    int p = threadIdx.x, c = blockIdx.x;          // blockDim = kP
    float ar = lb[2 * p], ai = lb[2 * p + 1];
    float xr = 0.f, xi = 0.f;
    float* base = Bu + (size_t)c * kTCHUNK * k2P + 2 * p;
    for (int t = 0; t < kTCHUNK; ++t) {
        float2 b = *(const float2*)&base[(size_t)t * k2P];
        float nr = ar * xr - ai * xi + b.x;
        float ni = ar * xi + ai * xr + b.y;
        xr = nr; xi = ni;
        *(float2*)&base[(size_t)t * k2P] = make_float2(xr, xi);
    }
    *(float2*)&carry[(size_t)c * k2P + 2 * p] = make_float2(xr, xi);
}

__global__ void k_scan_carry(const float* __restrict__ lb, const float* __restrict__ carry,
                             float* __restrict__ pref) {
    int p = threadIdx.x;                          // single block, blockDim = kP
    float ar = lb[2 * p], ai = lb[2 * p + 1];
    float tr = ar, ti = ai;                       // a^64 via 6 complex squarings
#pragma unroll
    for (int s = 0; s < 6; ++s) { float nr = tr * tr - ti * ti; ti = 2.f * tr * ti; tr = nr; }
    float pr = 0.f, pi = 0.f;
    for (int c = 0; c < kNCHUNK; ++c) {
        *(float2*)&pref[(size_t)c * k2P + 2 * p] = make_float2(pr, pi);
        float2 cv = *(const float2*)&carry[(size_t)c * k2P + 2 * p];
        float nr = tr * pr - ti * pi + cv.x;
        float ni = tr * pi + ti * pr + cv.y;
        pr = nr; pi = ni;
    }
}

__global__ void k_scan_fix(const float* __restrict__ lb, const float* __restrict__ Bu,
                           const float* __restrict__ pref, __bf16* __restrict__ xs) {
    int p = threadIdx.x, c = blockIdx.x;
    float ar = lb[2 * p], ai = lb[2 * p + 1];
    float2 cv = *(const float2*)&pref[(size_t)c * k2P + 2 * p];
    float pwr = ar, pwi = ai;
    const float* base = Bu + (size_t)c * kTCHUNK * k2P + 2 * p;
    __bf16* xbase = xs + (size_t)c * kTCHUNK * k2P + 2 * p;
    for (int t = 0; t < kTCHUNK; ++t) {
        float2 lv = *(const float2*)&base[(size_t)t * k2P];
        float xr = lv.x + pwr * cv.x - pwi * cv.y;
        float xi = lv.y + pwr * cv.y + pwi * cv.x;
        xbase[(size_t)t * k2P + 0] = (__bf16)xr;
        xbase[(size_t)t * k2P + 1] = (__bf16)xi;
        float nr = pwr * ar - pwi * ai; pwi = pwr * ai + pwi * ar; pwr = nr;
    }
}

// ==================== GEMM-2: out = xs(bf16) @ W2t^T + D⊙u -> f32 (L x H) ====================
__global__ void __launch_bounds__(NTHR)
k_gemm2(const __bf16* __restrict__ Ax, const __bf16* __restrict__ Bw,
        const float* __restrict__ U, const float* __restrict__ Dv,
        float* __restrict__ Cout) {
    __shared__ __bf16 As[2][BM][LDSP];
    __shared__ __bf16 Bs[2][BN][LDSP];
    const int tid = threadIdx.x, lane = tid & 31, wave = tid >> 5;
    const int half = lane >> 4, l15 = lane & 15;
    const int m0 = blockIdx.y * BM, n0 = blockIdx.x * BN;
    constexpr int K = k2P, N = kH, nK = K / BK;

    // B async staging (Bw is (H x 2P), K-contiguous)
    const int bn = tid >> 2, bc = (tid & 3) * 8;
    const __bf16* bsrc = Bw + (size_t)(n0 + bn) * K + bc;
    const unsigned bdst[2] = { lds_off(&Bs[0][bn][bc]), lds_off(&Bs[1][bn][bc]) };

    // A async staging: 2 x b128 per thread (Ax is bf16, K-contiguous)
    const int r1 = tid >> 2,          c1 = (tid & 3) * 8;
    const int r2 = (tid + NTHR) >> 2, c2 = ((tid + NTHR) & 3) * 8;
    const __bf16* asrc1 = Ax + (size_t)(m0 + r1) * K + c1;
    const __bf16* asrc2 = Ax + (size_t)(m0 + r2) * K + c2;
    const unsigned adst1[2] = { lds_off(&As[0][r1][c1]), lds_off(&As[1][r1][c1]) };
    const unsigned adst2[2] = { lds_off(&As[0][r2][c2]), lds_off(&As[1][r2][c2]) };

    auto issue_tile = [&](int k0, int buf) {
        async_ld_b128(adst1[buf], asrc1 + k0);
        async_ld_b128(adst2[buf], asrc2 + k0);
        async_ld_b128(bdst[buf],  bsrc  + k0);
    };

    v8f acc[4];
#pragma unroll
    for (int ns = 0; ns < 4; ++ns) acc[ns] = (v8f){0.f,0.f,0.f,0.f,0.f,0.f,0.f,0.f};

    issue_tile(0, 0);
    wait_async0();
    __syncthreads();

    for (int kt = 0; kt < nK; ++kt) {
        const int buf = kt & 1;
        if (kt + 1 < nK) issue_tile((kt + 1) * BK, buf ^ 1);
        v16bf af = load_afrag(&As[buf][16 * wave + l15][0], half);
#pragma unroll
        for (int ns = 0; ns < 4; ++ns) {
            v16bf bf_ = load_bfrag(&Bs[buf][16 * ns + l15][0], half);
            acc[ns] = __builtin_amdgcn_wmma_f32_16x16x32_bf16(
                false, af, false, bf_, (short)0, acc[ns], false, false);
        }
        if (kt + 1 < nK) wait_async0();
        __syncthreads();
    }
#pragma unroll
    for (int ns = 0; ns < 4; ++ns)
#pragma unroll
        for (int i = 0; i < 8; ++i) {
            int r = m0 + 16 * wave + 8 * half + i;
            int c = n0 + 16 * ns + l15;
            Cout[(size_t)r * N + c] = acc[ns][i] + Dv[c] * U[(size_t)r * N + c];
        }
}

// ==================== host launcher ====================
extern "C" void kernel_launch(void* const* d_in, const int* in_sizes, int n_in,
                              void* d_out, int out_size, void* d_ws, size_t ws_size,
                              hipStream_t stream) {
    const float* u        = (const float*)d_in[0];   // (L,H)
    const float* Lre      = (const float*)d_in[1];   // (P,)
    const float* Lim      = (const float*)d_in[2];   // (P,)
    const float* Bt       = (const float*)d_in[3];   // (P,H,2)
    const float* Ct       = (const float*)d_in[4];   // (H,P,2)
    const float* Dv       = (const float*)d_in[5];   // (H,)
    const float* log_step = (const float*)d_in[6];   // (P,1)
    float* out = (float*)d_out;

    char* ws = (char*)d_ws;
    auto alignup = [](size_t x) { return (x + 255) & ~(size_t)255; };
    size_t off = 0;
    float*  lb    = (float*)(ws + off);  off = alignup(off + (size_t)k2P * 4);
    __bf16* W1t   = (__bf16*)(ws + off); off = alignup(off + (size_t)k2P * kH * 2);
    __bf16* W2t   = (__bf16*)(ws + off); off = alignup(off + (size_t)kH * k2P * 2);
    float*  Bu    = (float*)(ws + off);  off = alignup(off + (size_t)kL * k2P * 4);
    float*  carry = (float*)(ws + off);  off = alignup(off + (size_t)kNCHUNK * k2P * 4);
    float*  pref  = (float*)(ws + off);  off = alignup(off + (size_t)kNCHUNK * k2P * 4);
    __bf16* xs    = (__bf16*)(ws + off); off = alignup(off + (size_t)kL * k2P * 2);

    k_prep_lambda<<<1, kP, 0, stream>>>(Lre, Lim, log_step, lb);
    k_prep_W1<<<(kP * kH) / 256, 256, 0, stream>>>(Lre, Lim, Bt, lb, W1t);
    k_prep_W2<<<(kP * kH) / 256, 256, 0, stream>>>(Ct, W2t);

    k_gemm1<<<dim3(k2P / BN, kL / BM), NTHR, 0, stream>>>(u, W1t, Bu);

    k_scan_local<<<kNCHUNK, kP, 0, stream>>>(lb, Bu, carry);
    k_scan_carry<<<1, kP, 0, stream>>>(lb, carry, pref);
    k_scan_fix<<<kNCHUNK, kP, 0, stream>>>(lb, Bu, pref, xs);

    k_gemm2<<<dim3(kH / BN, kL / BM), NTHR, 0, stream>>>(xs, W2t, u, Dv, out);
}
